// ParallelMLPs_76124000354877
// MI455X (gfx1250) — matrix operand order repaired and checked by reading, not compiled, not run on hardware
//
#include <hip/hip_runtime.h>
#include <hip/hip_bf16.h>

typedef __attribute__((ext_vector_type(16))) __bf16 v16bf;
typedef __attribute__((ext_vector_type(8)))  __bf16 v8bf;
typedef __attribute__((ext_vector_type(4)))  __bf16 v4bf;
typedef __attribute__((ext_vector_type(8)))  float  v8f;
typedef __attribute__((ext_vector_type(4)))  float  fx4;

#define NI       64
#define T_TOT    4096
#define DIN      128
#define DH       256
#define DOUT     128
#define TT       128          // token tile per workgroup
#define XS       136          // padded LDS row stride (elems) for 128-wide rows; 272B, 16B-aligned
#define HS       264          // padded LDS row stride (elems) for 256-wide rows; 528B, 16B-aligned

// LDS layout (bf16 element offsets). H aliases W1T after GEMM1 (barrier-protected).
#define LDS_W1T   0                      // [256][XS]  -> 34816 elems
#define LDS_X     (DH * XS)              // [128][XS]  -> 17408 elems
#define LDS_W2T   (LDS_X + TT * XS)      // [128][HS]  -> 33792 elems
#define LDS_ELEMS (LDS_W2T + DOUT * HS)  // 86016 elems
#define LDS_BYTES (LDS_ELEMS * 2)        // 172032 B  (<= 320KB WGP LDS)

__device__ static inline void store_pk4(__bf16* dst, float a, float b, float c, float d) {
  *(v4bf*)dst = (v4bf){(__bf16)a, (__bf16)b, (__bf16)c, (__bf16)d};  // one ds_store_b64
}

// A fragment (16x32 bf16): lane l<16 holds row M=l, K = {k0..k0+7, k0+16..k0+23};
// lane l>=16 holds row M=l-16, K = {k0+8..k0+15, k0+24..k0+31}. p already includes hi*8.
__device__ static inline v16bf frag_a(const __bf16* p) {
  v8bf lo = *(const v8bf*)p;
  v8bf hi = *(const v8bf*)(p + 16);
  return __builtin_shufflevector(lo, hi, 0,1,2,3,4,5,6,7,8,9,10,11,12,13,14,15);
}
// B fragment (32x16 bf16): lanes 0-15 hold K={k0..k0+7, k0+16..k0+23} of column N=lane,
// lanes 16-31 hold K={k0+8..k0+15, k0+24..k0+31} of column N=lane-16. p already includes hi*8.
__device__ static inline v16bf frag_b(const __bf16* p) {
  v8bf lo = *(const v8bf*)p;
  v8bf hi = *(const v8bf*)(p + 16);
  return __builtin_shufflevector(lo, hi, 0,1,2,3,4,5,6,7,8,9,10,11,12,13,14,15);
}

#define WMMA_BF16(A, B, C) __builtin_amdgcn_wmma_f32_16x16x32_bf16( \
    false, (A), false, (B), (short)0, (C), false, false)

extern "C" __global__ void __launch_bounds__(256, 1)
parallel_mlp_wmma(const float* __restrict__ x,  const float* __restrict__ W1,
                  const float* __restrict__ b1, const float* __restrict__ W2,
                  const float* __restrict__ b2, float* __restrict__ out)
{
  extern __shared__ __bf16 lds[];
  __bf16* sW1T = lds + LDS_W1T;   // [n:256][k:128] stride XS
  __bf16* sX   = lds + LDS_X;     // [t:128][k:128] stride XS
  __bf16* sW2T = lds + LDS_W2T;   // [n:128][k:256] stride HS
  __bf16* sH   = lds + LDS_W1T;   // [t:128][h:256] stride HS (aliases W1T)

  const int tid  = threadIdx.x;
  const int inst = blockIdx.y;
  const int t0   = blockIdx.x * TT;

  const fx4*   xg  = (const fx4*)(x  + ((size_t)inst * T_TOT + t0) * DIN);
  const fx4*   w1g = (const fx4*)(W1 + (size_t)inst * DIN * DH);
  const fx4*   w2g = (const fx4*)(W2 + (size_t)inst * DH * DOUT);
  const float* b1g = b1 + inst * DH;
  const float* b2g = b2 + inst * DOUT;
  float*       og  = out + ((size_t)inst * T_TOT + t0) * DOUT;

  // ---- Stage x tile (128x128 f32 -> bf16), streamed (non-temporal) ----
  #pragma unroll
  for (int it = 0; it < (TT * DIN / 4) / 256; ++it) {     // 16 iters
    int idx = tid + it * 256;
    fx4 v = __builtin_nontemporal_load(&xg[idx]);
    int row = idx >> 5;
    int k4  = (idx & 31) << 2;
    store_pk4(&sX[row * XS + k4], v.x, v.y, v.z, v.w);
  }
  // ---- Stage W1 (128x256) transposed -> sW1T[n][k] via 4x4 micro-tiles ----
  #pragma unroll
  for (int it = 0; it < (DIN / 4) * (DH / 4) / 256; ++it) {  // 8 iters
    int t4  = tid + it * 256;
    int k4  = (t4 >> 6) << 2;
    int n4t = t4 & 63;
    fx4 r0 = w1g[(k4 + 0) * (DH / 4) + n4t];
    fx4 r1 = w1g[(k4 + 1) * (DH / 4) + n4t];
    fx4 r2 = w1g[(k4 + 2) * (DH / 4) + n4t];
    fx4 r3 = w1g[(k4 + 3) * (DH / 4) + n4t];
    int n = n4t << 2;
    store_pk4(&sW1T[(n + 0) * XS + k4], r0.x, r1.x, r2.x, r3.x);
    store_pk4(&sW1T[(n + 1) * XS + k4], r0.y, r1.y, r2.y, r3.y);
    store_pk4(&sW1T[(n + 2) * XS + k4], r0.z, r1.z, r2.z, r3.z);
    store_pk4(&sW1T[(n + 3) * XS + k4], r0.w, r1.w, r2.w, r3.w);
  }
  // ---- Stage W2 (256x128) transposed -> sW2T[n][k] via 4x4 micro-tiles ----
  #pragma unroll
  for (int it = 0; it < (DH / 4) * (DOUT / 4) / 256; ++it) { // 8 iters
    int t4  = tid + it * 256;
    int k4  = (t4 >> 5) << 2;
    int n4t = t4 & 31;
    fx4 r0 = w2g[(k4 + 0) * (DOUT / 4) + n4t];
    fx4 r1 = w2g[(k4 + 1) * (DOUT / 4) + n4t];
    fx4 r2 = w2g[(k4 + 2) * (DOUT / 4) + n4t];
    fx4 r3 = w2g[(k4 + 3) * (DOUT / 4) + n4t];
    int n = n4t << 2;
    store_pk4(&sW2T[(n + 0) * HS + k4], r0.x, r1.x, r2.x, r3.x);
    store_pk4(&sW2T[(n + 1) * HS + k4], r0.y, r1.y, r2.y, r3.y);
    store_pk4(&sW2T[(n + 2) * HS + k4], r0.z, r1.z, r2.z, r3.z);
    store_pk4(&sW2T[(n + 3) * HS + k4], r0.w, r1.w, r2.w, r3.w);
  }
  __syncthreads();

  const int wave = tid >> 5;
  const int lane = tid & 31;
  const int ln   = lane & 15;
  const int hi   = lane >> 4;
  const int mp   = wave >> 1;          // M-pair: rows 32*mp .. 32*mp+31 (2 M-tiles)
  const int nh   = wave & 1;           // N-half (GEMM1: 128 cols, GEMM2: 64 cols)

  // ================= GEMM1: H = X(128x128) @ W1(128x256) =================
  // wave computes 2 M-tiles x 8 N-tiles; fragments double-buffered in groups of (2A, 4B)
  v8f acc[2][8];
  #pragma unroll
  for (int mi = 0; mi < 2; ++mi)
    #pragma unroll
    for (int j = 0; j < 8; ++j) acc[mi][j] = (v8f){0,0,0,0,0,0,0,0};
  float bias1[8];
  #pragma unroll
  for (int j = 0; j < 8; ++j) bias1[j] = b1g[(nh * 8 + j) * 16 + ln];

  {
    const __bf16* aB = sX   + (mp * 32 + ln) * XS + hi * 8;     // + mi*16*XS + kt*32
    const __bf16* bB = sW1T + (nh * 128 + ln) * XS + hi * 8;    // + (g*4+nt)*16*XS + kt*32
    v16bf Af[2][2], Bf[2][4];
    #define LOAD_A1(buf, kt) { Af[buf][0] = frag_a(aB + (kt) * 32); \
                               Af[buf][1] = frag_a(aB + 16 * XS + (kt) * 32); }
    #define LOAD_B1(buf, kt, g) { _Pragma("unroll") \
        for (int q = 0; q < 4; ++q) \
          Bf[buf][q] = frag_b(bB + ((g) * 4 + q) * 16 * XS + (kt) * 32); }
    LOAD_A1(0, 0); LOAD_B1(0, 0, 0);
    #pragma unroll
    for (int kt = 0; kt < DIN / 32; ++kt) {               // 4 k-steps
      const int ca = kt & 1;
      #pragma unroll
      for (int g = 0; g < 2; ++g) {                       // two 4-wide N groups
        const int cb = (kt * 2 + g) & 1, nb = cb ^ 1;
        if (g == 0) { LOAD_B1(nb, kt, 1); }
        else if (kt + 1 < DIN / 32) { LOAD_A1(ca ^ 1, kt + 1); LOAD_B1(nb, kt + 1, 0); }
        #pragma unroll
        for (int mi = 0; mi < 2; ++mi)
          #pragma unroll
          for (int nt = 0; nt < 4; ++nt)
            acc[mi][g * 4 + nt] = WMMA_BF16(Af[ca][mi], Bf[cb][nt], acc[mi][g * 4 + nt]);
      }
    }
    #undef LOAD_A1
    #undef LOAD_B1
  }
  __syncthreads();   // all reads of W1T/X done before H overwrites W1T

  // ---- bias + SiLU (fp32), write H as bf16 ----
  #pragma unroll
  for (int mi = 0; mi < 2; ++mi)
    #pragma unroll
    for (int j = 0; j < 8; ++j) {
      int col = (nh * 8 + j) * 16 + ln;
      #pragma unroll
      for (int r = 0; r < 8; ++r) {
        float v = acc[mi][j][r] + bias1[j];               // C layout: row = r + hi*8
        v = v * __fdividef(1.f, 1.f + __expf(-v));        // silu
        sH[(mp * 32 + mi * 16 + r + hi * 8) * HS + col] = (__bf16)v;
      }
    }
  __syncthreads();

  // ================= GEMM2: OUT = H(128x256) @ W2(256x128) =================
  // wave computes 2 M-tiles x 4 N-tiles; full (2A,4B) set double-buffered across kt
  v8f acc2[2][4];
  #pragma unroll
  for (int mi = 0; mi < 2; ++mi)
    #pragma unroll
    for (int j = 0; j < 4; ++j) acc2[mi][j] = (v8f){0,0,0,0,0,0,0,0};
  float bias2[4];
  #pragma unroll
  for (int j = 0; j < 4; ++j) bias2[j] = b2g[(nh * 4 + j) * 16 + ln];

  {
    const __bf16* aB = sH   + (mp * 32 + ln) * HS + hi * 8;
    const __bf16* bB = sW2T + (nh * 64 + ln) * HS + hi * 8;
    v16bf Af[2][2], Bf[2][4];
    #define LOAD_AB2(buf, kt) { \
        Af[buf][0] = frag_a(aB + (kt) * 32); \
        Af[buf][1] = frag_a(aB + 16 * HS + (kt) * 32); \
        _Pragma("unroll") \
        for (int q = 0; q < 4; ++q) \
          Bf[buf][q] = frag_b(bB + q * 16 * HS + (kt) * 32); }
    LOAD_AB2(0, 0);
    #pragma unroll
    for (int kt = 0; kt < DH / 32; ++kt) {                // 8 k-steps
      const int cur = kt & 1;
      if (kt + 1 < DH / 32) { LOAD_AB2(cur ^ 1, kt + 1); }
      #pragma unroll
      for (int mi = 0; mi < 2; ++mi)
        #pragma unroll
        for (int nt = 0; nt < 4; ++nt)
          acc2[mi][nt] = WMMA_BF16(Af[cur][mi], Bf[cur][nt], acc2[mi][nt]);
    }
    #undef LOAD_AB2
  }

  // ---- bias + fp32 non-temporal store ----
  #pragma unroll
  for (int mi = 0; mi < 2; ++mi)
    #pragma unroll
    for (int nt = 0; nt < 4; ++nt) {
      int col = (nh * 4 + nt) * 16 + ln;
      #pragma unroll
      for (int r = 0; r < 8; ++r) {
        float v = acc2[mi][nt][r] + bias2[nt];
        __builtin_nontemporal_store(
            v, &og[(size_t)(mp * 32 + mi * 16 + r + hi * 8) * DOUT + col]);
      }
    }
}

extern "C" void kernel_launch(void* const* d_in, const int* in_sizes, int n_in,
                              void* d_out, int out_size, void* d_ws, size_t ws_size,
                              hipStream_t stream) {
  (void)in_sizes; (void)n_in; (void)out_size; (void)d_ws; (void)ws_size;
  const float* x  = (const float*)d_in[0];
  const float* W1 = (const float*)d_in[1];
  const float* b1 = (const float*)d_in[2];
  const float* W2 = (const float*)d_in[3];
  const float* b2 = (const float*)d_in[4];
  float* out = (float*)d_out;

  hipFuncSetAttribute(reinterpret_cast<const void*>(parallel_mlp_wmma),
                      hipFuncAttributeMaxDynamicSharedMemorySize, LDS_BYTES);

  dim3 grid(T_TOT / TT, NI);   // 32 token-tiles x 64 instances = 2048 workgroups
  parallel_mlp_wmma<<<grid, 256, LDS_BYTES, stream>>>(x, W1, b1, W2, b2, out);
}
